// RBFN_18116172055027
// MI455X (gfx1250) — matrix operand order repaired and checked by reading, not compiled, not run on hardware
//
#include <hip/hip_runtime.h>

typedef __attribute__((ext_vector_type(16))) _Float16 v16h;
typedef __attribute__((ext_vector_type(8)))  float    v8f;
typedef __attribute__((ext_vector_type(2)))  __fp16   h2;   // cvt_pkrtz return type

#define THREADS        256
#define ROWS_PER_BLOCK 256   // 8 waves x 32 rows
#define TILE_C         256   // centres staged per pass (2 passes for C=512)
#define D_DIM          64

__global__ __launch_bounds__(THREADS) void rbfn_wmma_kernel(
    const float* __restrict__ x,        // (N, 64)
    const float* __restrict__ centres,  // (512, 64)
    const float* __restrict__ sigmas,   // (512)
    const float* __restrict__ W,        // (1, 512)
    const float* __restrict__ b,        // (1)
    float* __restrict__ out,            // (N, 1)
    int C)
{
    __shared__ _Float16 cf16[TILE_C * D_DIM];   // 32 KB: centres tile as f16
    __shared__ float    c2s[TILE_C];            // ||c||^2
    __shared__ float    s2s[TILE_C];            // sigma^2
    __shared__ float    wls[TILE_C];            // W row
    __shared__ float    x2s[ROWS_PER_BLOCK];    // ||x||^2 for block rows

    const int tid  = threadIdx.x;
    const int lane = tid & 31;
    const int wave = tid >> 5;
    const int half = lane >> 4;   // 0: lanes 0-15, 1: lanes 16-31
    const int l16  = lane & 15;

    const int blockRow = blockIdx.x * ROWS_PER_BLOCK;
    const int rowBase  = blockRow + wave * 32;

    // ---- per-row ||x||^2 for this block (one row per thread) ----
    {
        const float* xr = x + (size_t)(blockRow + tid) * D_DIM;
        float s = 0.f;
        #pragma unroll
        for (int k = 0; k < D_DIM; k += 4) {
            float4 v = *(const float4*)(xr + k);
            s += v.x * v.x + v.y * v.y + v.z * v.z + v.w * v.w;
        }
        x2s[tid] = s;
    }

    // ---- A fragments: 2 M-tiles x 2 K-chunks, loaded once, reused for all C ----
    // 16-bit A 16x32 layout: lanes 0-15 row=l16 hold K {kb..kb+7, kb+16..kb+23}
    // with kb=0; lanes 16-31 same rows with kb=8.
    v16h afrag[2][2];
    #pragma unroll
    for (int mt = 0; mt < 2; ++mt) {
        const float* xr = x + (size_t)(rowBase + mt * 16 + l16) * D_DIM;
        #pragma unroll
        for (int q = 0; q < 2; ++q) {
            const int kb = q * 32 + half * 8;
            v16h a;
            #pragma unroll
            for (int g = 0; g < 2; ++g) {            // groups at +0 and +16
                float4 f0 = *(const float4*)(xr + kb + g * 16);
                float4 f1 = *(const float4*)(xr + kb + g * 16 + 4);
                h2 p0 = __builtin_amdgcn_cvt_pkrtz(f0.x, f0.y);
                h2 p1 = __builtin_amdgcn_cvt_pkrtz(f0.z, f0.w);
                h2 p2 = __builtin_amdgcn_cvt_pkrtz(f1.x, f1.y);
                h2 p3 = __builtin_amdgcn_cvt_pkrtz(f1.z, f1.w);
                a[g * 8 + 0] = (_Float16)p0[0]; a[g * 8 + 1] = (_Float16)p0[1];
                a[g * 8 + 2] = (_Float16)p1[0]; a[g * 8 + 3] = (_Float16)p1[1];
                a[g * 8 + 4] = (_Float16)p2[0]; a[g * 8 + 5] = (_Float16)p2[1];
                a[g * 8 + 6] = (_Float16)p3[0]; a[g * 8 + 7] = (_Float16)p3[1];
            }
            afrag[mt][q] = a;
        }
    }

    __syncthreads();

    // x2 values this lane needs in the epilogue (row = rowBase + mt*16 + v + 8*half)
    float x2v[16];
    #pragma unroll
    for (int mt = 0; mt < 2; ++mt)
        #pragma unroll
        for (int v = 0; v < 8; ++v)
            x2v[mt * 8 + v] = x2s[wave * 32 + mt * 16 + v + 8 * half];

    float partial[16];
    #pragma unroll
    for (int i = 0; i < 16; ++i) partial[i] = 0.f;

    const int passes = C / TILE_C;   // 2
    for (int pass = 0; pass < passes; ++pass) {
        __syncthreads();             // previous pass done reading LDS

        // ---- stage TILE_C centres as f16 + c^2, sigma^2, W (one centre per thread) ----
        {
            const int c = pass * TILE_C + tid;
            const float* cr = centres + (size_t)c * D_DIM;
            float s = 0.f;
            #pragma unroll
            for (int k = 0; k < D_DIM; k += 4) {
                float4 v = *(const float4*)(cr + k);
                s += v.x * v.x + v.y * v.y + v.z * v.z + v.w * v.w;
                h2 p0 = __builtin_amdgcn_cvt_pkrtz(v.x, v.y);
                h2 p1 = __builtin_amdgcn_cvt_pkrtz(v.z, v.w);
                cf16[tid * D_DIM + k + 0] = (_Float16)p0[0];
                cf16[tid * D_DIM + k + 1] = (_Float16)p0[1];
                cf16[tid * D_DIM + k + 2] = (_Float16)p1[0];
                cf16[tid * D_DIM + k + 3] = (_Float16)p1[1];
            }
            c2s[tid] = s;
            float sg = sigmas[c];
            s2s[tid] = sg * sg;
            wls[tid] = W[c];
        }
        __syncthreads();

        // ---- sweep 16 column tiles of this pass ----
        for (int ct = 0; ct < TILE_C / 16; ++ct) {
            const int col = ct * 16 + l16;

            // B fragment (f16 32x16): lanes 0-15 col=l16 hold K {kb..kb+7, kb+16..kb+23} (kb=q*32),
            // lanes 16-31 same with kb+8.
            v16h bfrag[2];
            #pragma unroll
            for (int q = 0; q < 2; ++q) {
                const _Float16* p = &cf16[col * D_DIM + q * 32 + half * 8];
                v16h bb;
                #pragma unroll
                for (int i = 0; i < 16; ++i) bb[i] = p[i < 8 ? i : i + 8];
                bfrag[q] = bb;
            }

            const float c2c = c2s[col];
            const float s2c = s2s[col];
            const float wc  = wls[col];

            #pragma unroll
            for (int mt = 0; mt < 2; ++mt) {
                v8f acc = {};
                acc = __builtin_amdgcn_wmma_f32_16x16x32_f16(
                    false, afrag[mt][0], false, bfrag[0], (short)0, acc, false, false);
                acc = __builtin_amdgcn_wmma_f32_16x16x32_f16(
                    false, afrag[mt][1], false, bfrag[1], (short)0, acc, false, false);

                #pragma unroll
                for (int v = 0; v < 8; ++v) {
                    float sq = x2v[mt * 8 + v] + c2c - 2.0f * acc[v];
                    sq = fmaxf(sq, 0.0f);
                    partial[mt * 8 + v] += __expf(-sq * s2c) * wc;
                }
            }
        }
    }

    // ---- reduce over the 16 columns held across lanes of each half ----
    #pragma unroll
    for (int i = 0; i < 16; ++i) {
        float s = partial[i];
        s += __shfl_xor(s, 1, 32);
        s += __shfl_xor(s, 2, 32);
        s += __shfl_xor(s, 4, 32);
        s += __shfl_xor(s, 8, 32);
        partial[i] = s;
    }

    const float b0 = b[0];
    if (l16 == 0) {
        #pragma unroll
        for (int mt = 0; mt < 2; ++mt)
            #pragma unroll
            for (int v = 0; v < 8; ++v) {
                const int row = rowBase + mt * 16 + v + 8 * half;
                out[row] = partial[mt * 8 + v] + b0;
            }
    }
}

extern "C" void kernel_launch(void* const* d_in, const int* in_sizes, int n_in,
                              void* d_out, int out_size, void* d_ws, size_t ws_size,
                              hipStream_t stream) {
    const float* x       = (const float*)d_in[0];
    const float* centres = (const float*)d_in[1];
    const float* sigmas  = (const float*)d_in[2];
    const float* W       = (const float*)d_in[3];
    const float* b       = (const float*)d_in[4];
    float* out = (float*)d_out;

    const int N = in_sizes[0] / D_DIM;     // 131072
    const int C = in_sizes[1] / D_DIM;     // 512

    const int blocks = N / ROWS_PER_BLOCK; // 512
    rbfn_wmma_kernel<<<blocks, THREADS, 0, stream>>>(x, centres, sigmas, W, b, out, C);
}
